// HydraBlock_7765300871301
// MI455X (gfx1250) — matrix operand order repaired and checked by reading, not compiled, not run on hardware
//
#include <hip/hip_runtime.h>
#include <hip/hip_bf16.h>

// ---------------------------------------------------------------------------
// HydraBlock for MI455X (gfx1250): bf16 WMMA GEMM pipeline with async
// global->LDS staging (ASYNCcnt path) for A tiles and branch-free
// software-pipelined B-fragment loads (peeled-last-iteration rotation).
// B=4, T=2048, D=512, H=8, DH=64, DLAT=64, E=32. Tokens M = B*T = 8192.
// ---------------------------------------------------------------------------

typedef __attribute__((ext_vector_type(16))) __bf16 v16bf;
typedef __attribute__((ext_vector_type(8)))  __bf16 v8bf;
typedef __attribute__((ext_vector_type(8)))  float  v8f;

#define NTOK   8192
#define DMODEL 512
#define NHEAD  8
#define DHEAD  64
#define DLAT   64
#define NEXP   32
#define SEQT   2048
#define LDA_PAD 520   // 512 + 8 bf16 = 1040B row stride: 16B aligned, 4-bank skew

// -------------------- CDNA5 async global->LDS (16B per lane) ----------------
__device__ __forceinline__ void async_copy16(const void* gptr, unsigned lds_off) {
    asm volatile("global_load_async_to_lds_b128 %0, %1, off"
                 :: "v"(lds_off), "v"((unsigned long long)(uintptr_t)gptr)
                 : "memory");
}
__device__ __forceinline__ void wait_async0() {
    asm volatile("s_wait_asynccnt 0x0" ::: "memory");
}
__device__ __forceinline__ unsigned lds_addr_of(const void* p) {
    return (unsigned)(uintptr_t)p;   // low 32 bits of generic LDS addr = LDS offset
}

// -------------------- WMMA B fragment loader (ISA 7.12.2) -------------------
// B: 32x16 bf16, consumed column-major. Source is Bt (N x K row-major).
// lane<16: col = lane, K = 0..7,16..23 ; lane>=16: col = lane-16, K = 8..15,24..31
__device__ __forceinline__ v16bf load_fragB(const __bf16* __restrict__ pt, int ld) {
    int lane = threadIdx.x & 31;
    const __bf16* r = pt + (size_t)(lane & 15) * ld + ((lane & 16) ? 8 : 0);
    v8bf lo = *(const v8bf*)(r);
    v8bf hi = *(const v8bf*)(r + 16);
    v16bf f;
#pragma unroll
    for (int i = 0; i < 8; ++i) { f[i] = lo[i]; f[i + 8] = hi[i]; }
    return f;
}

// -------------------- weight prep: f32 (K,N) -> bf16 (N,K) ------------------
__global__ void transpose_to_bf16(const float* __restrict__ W, __bf16* __restrict__ Wt,
                                  int K, int N) {
    size_t batch = blockIdx.y;
    const float* w = W + batch * (size_t)K * N;
    __bf16* wt = Wt + batch * (size_t)K * N;
    size_t i = (size_t)blockIdx.x * blockDim.x + threadIdx.x;
    if (i >= (size_t)K * N) return;
    size_t k = i / N, n = i % N;
    wt[n * (size_t)K + k] = (__bf16)w[i];
}

// -------------------- rmsnorm (f32 in -> f32 and/or bf16 out) ---------------
__global__ void rmsnorm_kernel(const float* __restrict__ x, const float* __restrict__ gamma,
                               float* __restrict__ of, __bf16* __restrict__ ob) {
    int t = blockIdx.x;
    const float* xr = x + (size_t)t * DMODEL;
    __shared__ float red[8];
    float s = 0.f;
    for (int d = threadIdx.x; d < DMODEL; d += blockDim.x) { float v = xr[d]; s += v * v; }
#pragma unroll
    for (int off = 16; off > 0; off >>= 1) s += __shfl_down(s, off, 32);
    if ((threadIdx.x & 31) == 0) red[threadIdx.x >> 5] = s;
    __syncthreads();
    if (threadIdx.x == 0) {
        float tot = 0.f;
        for (int i = 0; i < (int)(blockDim.x >> 5); ++i) tot += red[i];
        red[0] = rsqrtf(tot / (float)DMODEL + 1e-6f);
    }
    __syncthreads();
    float inv = red[0];
    for (int d = threadIdx.x; d < DMODEL; d += blockDim.x) {
        float v = xr[d] * inv * gamma[d];
        if (of) of[(size_t)t * DMODEL + d] = v;
        if (ob) ob[(size_t)t * DMODEL + d] = (__bf16)v;
    }
}

// -------------------- generic bf16 WMMA GEMM (LDS-staged A) -----------------
// C[M,N] = A[M,K] @ Bt[N,K]^T + bias + residual.  K is a template constant so
// the pipelined inner loop is branch-free.  Block: 128 thr = 4 waves, tile
// 16 rows x 64 cols.  A tile (16 x K) async-staged to LDS once per block.
template<int K>
__global__ void gemm_bf16_wmma(const __bf16* __restrict__ A, const __bf16* __restrict__ Bt,
                               const float* __restrict__ bias, const float* __restrict__ residual,
                               float* __restrict__ Cf, __bf16* __restrict__ Cb, int N) {
    __shared__ __bf16 As[16 * LDA_PAD];
    int wave = threadIdx.x >> 5;
    int lane = threadIdx.x & 31;
    int m0 = blockIdx.y * 16;
    int n0 = blockIdx.x * 64 + wave * 16;

    {   // async-stage A: 16 rows x K bf16, 16B chunks
        const __bf16* Ag = A + (size_t)m0 * K;
        constexpr int kchunks = K / 8;
        for (int c = threadIdx.x; c < 16 * kchunks; c += 128) {
            int row = c / kchunks;
            int k8 = (c % kchunks) * 8;
            async_copy16(Ag + (size_t)row * K + k8,
                         lds_addr_of(&As[row * LDA_PAD + k8]));
        }
        wait_async0();
    }
    __syncthreads();
    if (n0 >= N) return;

    auto loadA = [&](int k0) -> v16bf {   // A frag from LDS (ds_load_b128 x2)
        const __bf16* p = &As[(lane & 15) * LDA_PAD + k0 + ((lane & 16) ? 8 : 0)];
        v8bf lo = *(const v8bf*)(p);
        v8bf hi = *(const v8bf*)(p + 16);
        v16bf f;
#pragma unroll
        for (int i = 0; i < 8; ++i) { f[i] = lo[i]; f[i + 8] = hi[i]; }
        return f;
    };

    const __bf16* Bb = Bt + (size_t)n0 * K;
    v8f c = {};
    v16bf bcur = load_fragB(Bb, K);
#pragma unroll 5
    for (int k0 = 0; k0 < K - 32; k0 += 32) {        // steady state: prefetch always
        v16bf bnext = load_fragB(Bb + k0 + 32, K);   // in flight over WMMA
        v16bf a = loadA(k0);
        c = __builtin_amdgcn_wmma_f32_16x16x32_bf16(false, a, false, bcur, (short)0, c,
                                                    false, false);
        bcur = bnext;
    }
    {   // peeled last step: no prefetch, no branch
        v16bf a = loadA(K - 32);
        c = __builtin_amdgcn_wmma_f32_16x16x32_bf16(false, a, false, bcur, (short)0, c,
                                                    false, false);
    }

    int col = n0 + (lane & 15);
    int rbase = m0 + ((lane & 16) ? 8 : 0);
    float bv = bias ? bias[col] : 0.f;
#pragma unroll
    for (int j = 0; j < 8; ++j) {
        size_t idx = (size_t)(rbase + j) * N + col;
        float v = c[j] + bv;
        if (residual) v += residual[idx];
        if (Cf) Cf[idx] = v;
        if (Cb) Cb[idx] = (__bf16)v;
    }
}

// -------------------- per-token attention (RoPE + 8x8 softmax + AV) ---------
__global__ void attn_kernel(const float* __restrict__ qg, const float* __restrict__ kvg,
                            __bf16* __restrict__ outb) {
    int t = blockIdx.x;          // token over B*T
    int pos = t % SEQT;          // sequence position for RoPE
    __shared__ float q[NHEAD][DHEAD], k[NHEAD][DHEAD], v[NHEAD][DHEAD], att[NHEAD][NHEAD];
    const float* qr = qg + (size_t)t * DMODEL;
    const float* kvr = kvg + (size_t)t * (2 * DMODEL);
    for (int i = threadIdx.x; i < NHEAD * DHEAD; i += 64) {
        int h = i >> 6, d = i & 63;
        q[h][d] = qr[i];
        k[h][d] = kvr[h * 128 + d];
        v[h][d] = kvr[h * 128 + 64 + d];
    }
    __syncthreads();
    for (int i = threadIdx.x; i < NHEAD * 32; i += 64) {   // rotate-half RoPE
        int h = i >> 5, d = i & 31;
        float inv_freq = __powf(10000.f, -(float)(2 * d) / 64.f);
        float sn, cs;
        __sincosf((float)pos * inv_freq, &sn, &cs);
        float q1 = q[h][d], q2 = q[h][d + 32];
        q[h][d] = q1 * cs - q2 * sn; q[h][d + 32] = q1 * sn + q2 * cs;
        float k1 = k[h][d], k2 = k[h][d + 32];
        k[h][d] = k1 * cs - k2 * sn; k[h][d + 32] = k1 * sn + k2 * cs;
    }
    __syncthreads();
    {   // scores: 64 threads, one (h,g) each
        int h = threadIdx.x >> 3, g = threadIdx.x & 7;
        float s = 0.f;
#pragma unroll
        for (int d = 0; d < DHEAD; ++d) s += q[h][d] * k[g][d];
        att[h][g] = s * 0.125f;  // 1/sqrt(64)
    }
    __syncthreads();
    if (threadIdx.x < NHEAD) {   // row softmax
        int h = threadIdx.x;
        float mx = att[h][0];
#pragma unroll
        for (int g = 1; g < NHEAD; ++g) mx = fmaxf(mx, att[h][g]);
        float e[NHEAD], sum = 0.f;
#pragma unroll
        for (int g = 0; g < NHEAD; ++g) { e[g] = __expf(att[h][g] - mx); sum += e[g]; }
#pragma unroll
        for (int g = 0; g < NHEAD; ++g) att[h][g] = e[g] / sum;
    }
    __syncthreads();
    for (int i = threadIdx.x; i < NHEAD * DHEAD; i += 64) {
        int h = i >> 6, d = i & 63;
        float s = 0.f;
#pragma unroll
        for (int g = 0; g < NHEAD; ++g) s += att[h][g] * v[g][d];
        outb[(size_t)t * DMODEL + i] = (__bf16)s;
    }
}

// -------------------- gate: softmax(h2 @ wg + bg) ---------------------------
// 256 threads = 8 waves; one wave per token, one lane per expert.
__global__ void gate_kernel(const float* __restrict__ h2, const float* __restrict__ wg,
                            const float* __restrict__ bg, float* __restrict__ gate) {
    int tok = blockIdx.x * 8 + (threadIdx.x >> 5);
    int e = threadIdx.x & 31;
    const float* hr = h2 + (size_t)tok * DMODEL;
    float s = bg[e];
    for (int d = 0; d < DMODEL; ++d) s += hr[d] * wg[d * NEXP + e];   // coalesced over e
    float mx = s;
#pragma unroll
    for (int off = 16; off > 0; off >>= 1) mx = fmaxf(mx, __shfl_xor(mx, off, 32));
    float ex = __expf(s - mx);
    float sum = ex;
#pragma unroll
    for (int off = 16; off > 0; off >>= 1) sum += __shfl_xor(sum, off, 32);
    gate[(size_t)tok * NEXP + e] = ex / sum;
}

// -------------------- MoE: out = x2 + sum_e gate_e * (h2 @ W_e) -------------
// Block: 128 thr = 4 waves; tile 32 rows x 64 cols. A (32x512) async-staged
// to LDS once; each wave runs 32 experts x 16 K-steps with TWO 16x16
// accumulators sharing each software-pipelined B fragment (branch-free).
__global__ void moe_wmma_kernel(const __bf16* __restrict__ H2, const __bf16* __restrict__ EWt,
                                const float* __restrict__ gate, const float* __restrict__ x2,
                                float* __restrict__ out) {
    __shared__ __bf16 As[32 * LDA_PAD];
    __shared__ float gs[32][NEXP];
    int wave = threadIdx.x >> 5;
    int lane = threadIdx.x & 31;
    int m0 = blockIdx.y * 32;
    int n0 = blockIdx.x * 64 + wave * 16;

    {   // async-stage A: 32 rows x 512 bf16 = 32KB, 16B chunks (16 per thread)
        const __bf16* Ag = H2 + (size_t)m0 * DMODEL;
        for (int c = threadIdx.x; c < 32 * (DMODEL / 8); c += 128) {
            int row = c / (DMODEL / 8);
            int k8 = (c % (DMODEL / 8)) * 8;
            async_copy16(Ag + (size_t)row * DMODEL + k8,
                         lds_addr_of(&As[row * LDA_PAD + k8]));
        }
        for (int i = threadIdx.x; i < 32 * NEXP; i += 128)
            gs[i >> 5][i & 31] = gate[(size_t)(m0 + (i >> 5)) * NEXP + (i & 31)];
        wait_async0();
    }
    __syncthreads();

    auto loadA = [&](int rbase, int k0) -> v16bf {
        const __bf16* p = &As[(rbase + (lane & 15)) * LDA_PAD + k0 + ((lane & 16) ? 8 : 0)];
        v8bf lo = *(const v8bf*)(p);
        v8bf hi = *(const v8bf*)(p + 16);
        v16bf f;
#pragma unroll
        for (int i = 0; i < 8; ++i) { f[i] = lo[i]; f[i + 8] = hi[i]; }
        return f;
    };

    int col = n0 + (lane & 15);
    int rb = (lane & 16) ? 8 : 0;
    float acc0[8], acc1[8];
#pragma unroll
    for (int j = 0; j < 8; ++j) {
        acc0[j] = x2[(size_t)(m0 + rb + j) * DMODEL + col];
        acc1[j] = x2[(size_t)(m0 + 16 + rb + j) * DMODEL + col];
    }

    const __bf16* Bbase = EWt + (size_t)n0 * DMODEL;
    for (int e = 0; e < NEXP; ++e) {
        const __bf16* Bb = Bbase + (size_t)e * DMODEL * DMODEL;
        v8f c0 = {}, c1 = {};
        v16bf bcur = load_fragB(Bb, DMODEL);
#pragma unroll 3
        for (int k0 = 0; k0 < DMODEL - 32; k0 += 32) {   // steady state: prefetch always
            v16bf bnext = load_fragB(Bb + k0 + 32, DMODEL);
            v16bf a0 = loadA(0, k0);
            v16bf a1 = loadA(16, k0);
            c0 = __builtin_amdgcn_wmma_f32_16x16x32_bf16(false, a0, false, bcur, (short)0, c0,
                                                         false, false);
            c1 = __builtin_amdgcn_wmma_f32_16x16x32_bf16(false, a1, false, bcur, (short)0, c1,
                                                         false, false);
            bcur = bnext;
        }
        {   // peeled last step
            v16bf a0 = loadA(0, DMODEL - 32);
            v16bf a1 = loadA(16, DMODEL - 32);
            c0 = __builtin_amdgcn_wmma_f32_16x16x32_bf16(false, a0, false, bcur, (short)0, c0,
                                                         false, false);
            c1 = __builtin_amdgcn_wmma_f32_16x16x32_bf16(false, a1, false, bcur, (short)0, c1,
                                                         false, false);
        }
#pragma unroll
        for (int j = 0; j < 8; ++j) {
            acc0[j] += gs[rb + j][e] * c0[j];
            acc1[j] += gs[16 + rb + j][e] * c1[j];
        }
    }
#pragma unroll
    for (int j = 0; j < 8; ++j) {
        out[(size_t)(m0 + rb + j) * DMODEL + col] = acc0[j];
        out[(size_t)(m0 + 16 + rb + j) * DMODEL + col] = acc1[j];
    }
}

// ---------------------------------------------------------------------------
static inline size_t al256(size_t x) { return (x + 255) & ~(size_t)255; }

extern "C" void kernel_launch(void* const* d_in, const int* in_sizes, int n_in,
                              void* d_out, int out_size, void* d_ws, size_t ws_size,
                              hipStream_t stream) {
    const float* x      = (const float*)d_in[0];
    const float* gamma1 = (const float*)d_in[1];
    const float* gamma2 = (const float*)d_in[2];
    const float* wq     = (const float*)d_in[3];
    const float* bq     = (const float*)d_in[4];
    const float* wkd    = (const float*)d_in[5];
    const float* bkd    = (const float*)d_in[6];
    const float* wku    = (const float*)d_in[7];
    const float* bku    = (const float*)d_in[8];
    const float* wo     = (const float*)d_in[9];
    const float* bo     = (const float*)d_in[10];
    const float* wg     = (const float*)d_in[11];
    const float* bg     = (const float*)d_in[12];
    const float* ew     = (const float*)d_in[13];
    float* out = (float*)d_out;

    // ---- carve workspace ----
    char* p = (char*)d_ws;
    auto take = [&](size_t bytes) { char* r = p; p += al256(bytes); return r; };
    __bf16* h1b  = (__bf16*)take((size_t)NTOK * DMODEL * 2);
    float*  qf   = (float*) take((size_t)NTOK * DMODEL * 4);
    __bf16* kdb  = (__bf16*)take((size_t)NTOK * DLAT * 2);
    float*  kvf  = (float*) take((size_t)NTOK * 2 * DMODEL * 4);
    __bf16* attb = (__bf16*)take((size_t)NTOK * DMODEL * 2);
    float*  x2   = (float*) take((size_t)NTOK * DMODEL * 4);
    float*  h2f  = (float*) take((size_t)NTOK * DMODEL * 4);
    __bf16* h2b  = (__bf16*)take((size_t)NTOK * DMODEL * 2);
    float*  gate = (float*) take((size_t)NTOK * NEXP * 4);
    __bf16* wqT  = (__bf16*)take((size_t)DMODEL * DMODEL * 2);
    __bf16* wkdT = (__bf16*)take((size_t)DMODEL * DLAT * 2);
    __bf16* wkuT = (__bf16*)take((size_t)DLAT * 2 * DMODEL * 2);
    __bf16* woT  = (__bf16*)take((size_t)DMODEL * DMODEL * 2);
    __bf16* ewT  = (__bf16*)take((size_t)NEXP * DMODEL * DMODEL * 2);
    (void)ws_size; (void)in_sizes; (void)n_in; (void)out_size;

    // ---- weight prep: transpose + bf16 ----
    {
        int n = DMODEL * DMODEL;
        transpose_to_bf16<<<dim3((n + 255) / 256, 1), 256, 0, stream>>>(wq, wqT, DMODEL, DMODEL);
        transpose_to_bf16<<<dim3((DMODEL * DLAT + 255) / 256, 1), 256, 0, stream>>>(wkd, wkdT, DMODEL, DLAT);
        transpose_to_bf16<<<dim3((DLAT * 2 * DMODEL + 255) / 256, 1), 256, 0, stream>>>(wku, wkuT, DLAT, 2 * DMODEL);
        transpose_to_bf16<<<dim3((n + 255) / 256, 1), 256, 0, stream>>>(wo, woT, DMODEL, DMODEL);
        transpose_to_bf16<<<dim3((n + 255) / 256, NEXP), 256, 0, stream>>>(ew, ewT, DMODEL, DMODEL);
    }

    // ---- attention branch ----
    rmsnorm_kernel<<<NTOK, 128, 0, stream>>>(x, gamma1, nullptr, h1b);

    dim3 gemmBlk(128);
    // q = h @ wq + bq          (f32)
    gemm_bf16_wmma<DMODEL><<<dim3(DMODEL / 64, NTOK / 16), gemmBlk, 0, stream>>>(
        h1b, wqT, bq, nullptr, qf, nullptr, DMODEL);
    // kd = h @ wkd + bkd       (bf16)
    gemm_bf16_wmma<DMODEL><<<dim3(1, NTOK / 16), gemmBlk, 0, stream>>>(
        h1b, wkdT, bkd, nullptr, nullptr, kdb, DLAT);
    // kv = kd @ wku + bku      (f32)
    gemm_bf16_wmma<DLAT><<<dim3(2 * DMODEL / 64, NTOK / 16), gemmBlk, 0, stream>>>(
        kdb, wkuT, bku, nullptr, kvf, nullptr, 2 * DMODEL);
    // per-token RoPE + 8x8 attention
    attn_kernel<<<NTOK, 64, 0, stream>>>(qf, kvf, attb);
    // x2 = x + attout @ wo + bo
    gemm_bf16_wmma<DMODEL><<<dim3(DMODEL / 64, NTOK / 16), gemmBlk, 0, stream>>>(
        attb, woT, bo, x, x2, nullptr, DMODEL);

    // ---- MoE branch ----
    rmsnorm_kernel<<<NTOK, 128, 0, stream>>>(x2, gamma2, h2f, h2b);
    gate_kernel<<<NTOK / 8, 256, 0, stream>>>(h2f, wg, bg, gate);
    moe_wmma_kernel<<<dim3(DMODEL / 64, NTOK / 32), gemmBlk, 0, stream>>>(
        h2b, ewT, gate, x2, out);
}
